// DMNmodel_16243566313602
// MI455X (gfx1250) — compile-verified, matrix operand order checked by reading, not run on hardware
//
#include <hip/hip_runtime.h>
#include <hip/hip_bf16.h>

typedef __bf16 bf16_t;
typedef __attribute__((ext_vector_type(16))) __bf16 v16bf;
typedef __attribute__((ext_vector_type(8)))  __bf16 v8bf;
typedef __attribute__((ext_vector_type(8)))  float  v8f;

#define TM 128
#define TN 256
#define TK 32
#define LDS_STRIDE 40   // 32 payload + 8 pad bf16 -> 80B rows, 16B aligned, spreads banks

__device__ __forceinline__ float sigmoidf_(float x) { return 1.0f / (1.0f + __expf(-x)); }

// gfx1250 async copy: global memory -> LDS, 16B per lane, tracked by ASYNCcnt.
// VDST = per-lane LDS byte address (low 32 bits of shared pointer), VADDR = 64-bit global addr.
__device__ __forceinline__ void async_ld16(const bf16_t* g, bf16_t* l) {
    unsigned           ldsAddr = (unsigned)(uintptr_t)l;
    unsigned long long gAddr   = (unsigned long long)(uintptr_t)g;
    asm volatile("global_load_async_to_lds_b128 %0, %1, off"
                 :: "v"(ldsAddr), "v"(gAddr)
                 : "memory");
}
__device__ __forceinline__ void async_wait0() {
    asm volatile("s_wait_asynccnt 0x0" ::: "memory");
}

// Build a 16x32 bf16 WMMA fragment from LDS.
// Lane L (<16): row = L, K = 0..7 and 16..23 ; lane L (>=16): row = L-16, K = 8..15 and 24..31.
__device__ __forceinline__ v16bf frag_from_lds(const bf16_t* base, int row, int kbase) {
    const v8bf lo = *reinterpret_cast<const v8bf*>(base + row * LDS_STRIDE + kbase);
    const v8bf hi = *reinterpret_cast<const v8bf*>(base + row * LDS_STRIDE + kbase + 16);
    v16bf r;
#pragma unroll
    for (int i = 0; i < 8; ++i) { r[i] = lo[i]; r[i + 8] = hi[i]; }
    return r;
}

// ---------------------------------------------------------------------------
// Generic bf16 GEMM:  D[M][N] (f32) = A[M][K](bf16,row-major) * W[N][K](bf16) (+bias) (+=D)
// Grid (N/256, M/128), 256 threads = 8 waves; wave w -> rows (w&1)*64, cols (w>>1)*64.
// Double-buffered LDS filled by global_load_async_to_lds_b128; 16 WMMA per wave per K-step.
// ---------------------------------------------------------------------------
__global__ __launch_bounds__(256) void gemm_bf16_kernel(
    float* __restrict__ D, const bf16_t* __restrict__ A, const bf16_t* __restrict__ W,
    const float* __restrict__ bias, int N, int K, int ldw, int wcol, int accflag)
{
    __shared__ bf16_t lsA[2][TM * LDS_STRIDE];
    __shared__ bf16_t lsB[2][TN * LDS_STRIDE];
    const int tid  = threadIdx.x;
    const int lane = tid & 31;
    const int wave = tid >> 5;
    const int wm   = wave & 1;
    const int wn   = wave >> 1;
    const long tileM = (long)blockIdx.y * TM;
    const int  tileN = blockIdx.x * TN;

    v8f acc[4][4];
#pragma unroll
    for (int mi = 0; mi < 4; ++mi)
#pragma unroll
        for (int ni = 0; ni < 4; ++ni)
#pragma unroll
            for (int e = 0; e < 8; ++e) acc[mi][ni][e] = 0.0f;

    auto stage = [&](int k0, int buf) {
#pragma unroll
        for (int i = 0; i < 2; ++i) {           // A tile: 128 rows x 64B
            int idx = tid + i * 256;
            int row = idx >> 2;
            int c8  = (idx & 3) * 8;
            async_ld16(&A[(size_t)(tileM + row) * K + k0 + c8],
                       &lsA[buf][row * LDS_STRIDE + c8]);
        }
#pragma unroll
        for (int i = 0; i < 4; ++i) {           // B tile: 256 rows x 64B
            int idx = tid + i * 256;
            int row = idx >> 2;
            int c8  = (idx & 3) * 8;
            async_ld16(&W[(size_t)(tileN + row) * ldw + wcol + k0 + c8],
                       &lsB[buf][row * LDS_STRIDE + c8]);
        }
    };

    stage(0, 0);
    async_wait0();
    __syncthreads();

    int cur = 0;
    for (int k0 = 0; k0 < K; k0 += TK) {
        if (k0 + TK < K) stage(k0 + TK, cur ^ 1);   // prefetch next tile (async)

        const int kb = (lane & 16) ? 8 : 0;
        const int rr = lane & 15;
        v16bf af[4], wf[4];
#pragma unroll
        for (int mi = 0; mi < 4; ++mi) af[mi] = frag_from_lds(lsA[cur], wm * 64 + mi * 16 + rr, kb);
#pragma unroll
        for (int ni = 0; ni < 4; ++ni) wf[ni] = frag_from_lds(lsB[cur], wn * 64 + ni * 16 + rr, kb);
#pragma unroll
        for (int mi = 0; mi < 4; ++mi)
#pragma unroll
            for (int ni = 0; ni < 4; ++ni)
                acc[mi][ni] = __builtin_amdgcn_wmma_f32_16x16x32_bf16(
                    false, af[mi], false, wf[ni], (short)0, acc[mi][ni], false, false);

        async_wait0();        // next tile landed in LDS
        __syncthreads();      // published to all waves; safe to overwrite 'cur' next step
        cur ^= 1;
    }

    const int nlo = lane & 15;
    const int mhi = (lane >> 4) << 3;
#pragma unroll
    for (int mi = 0; mi < 4; ++mi) {
#pragma unroll
        for (int ni = 0; ni < 4; ++ni) {
            int   n    = tileN + wn * 64 + ni * 16 + nlo;
            float bv   = bias ? bias[n] : 0.0f;
            long mbase = tileM + wm * 64 + mi * 16 + mhi;
#pragma unroll
            for (int r = 0; r < 8; ++r) {
                size_t o = (size_t)(mbase + r) * N + n;
                float  v = acc[mi][ni][r] + bv;
                if (accflag) v += D[o];
                D[o] = v;
            }
        }
    }
}

// ---------------------------------------------------------------------------
// Fused gate GEMM. Rows = (b*S+s), N = H, virtual K = 5H.
// A-blocks generated on the fly from C: j=0:C, 1:C*R, 2:C*P, 3:|C-R|, 4:|C-P|
// matching gW1 column blocks {0,3H,4H,5H,6H}. B streamed via async-to-LDS.
// Epilogue adds base + sR*w8 + sP*w9, tanh -> Hid (bf16).
// ---------------------------------------------------------------------------
__global__ __launch_bounds__(256) void fused_gate_gemm_kernel(
    bf16_t* __restrict__ Hid, const bf16_t* __restrict__ Cbf,
    const float* __restrict__ Rv,  // Mcur (B,H)
    const float* __restrict__ Pv,  // Q    (B,H)
    const bf16_t* __restrict__ gW1p,  // (H rows, 7H cols) bf16
    const float* __restrict__ w8, const float* __restrict__ w9,
    const float* __restrict__ basev,  // (B,H)
    const float* __restrict__ sR, const float* __restrict__ sP,  // (B*S)
    int S, int H)
{
    __shared__ bf16_t lsA[2][TM * LDS_STRIDE];
    __shared__ bf16_t lsB[2][TN * LDS_STRIDE];
    const int tid  = threadIdx.x;
    const int lane = tid & 31;
    const int wave = tid >> 5;
    const int wm   = wave & 1;
    const int wn   = wave >> 1;
    const long tileM = (long)blockIdx.y * TM;
    const int  tileN = blockIdx.x * TN;
    const int  b     = (int)(tileM / S);      // 128-row tile stays inside one batch (S=256)
    const float* Rb  = Rv + (size_t)b * H;
    const float* Pb  = Pv + (size_t)b * H;
    const int  H7    = 7 * H;

    v8f acc[4][4];
#pragma unroll
    for (int mi = 0; mi < 4; ++mi)
#pragma unroll
        for (int ni = 0; ni < 4; ++ni)
#pragma unroll
            for (int e = 0; e < 8; ++e) acc[mi][ni][e] = 0.0f;

    auto stageB = [&](int kg, int buf) {      // weights: async global->LDS
        int j    = kg / H;
        int kk   = kg - j * H;
        int coff = (j == 0) ? 0 : (j + 2) * H;
#pragma unroll
        for (int i = 0; i < 4; ++i) {
            int idx = tid + i * 256;
            int row = idx >> 2;
            int c8  = (idx & 3) * 8;
            async_ld16(&gW1p[(size_t)(tileN + row) * H7 + coff + kk + c8],
                       &lsB[buf][row * LDS_STRIDE + c8]);
        }
    };
    auto stageA = [&](int kg, int buf) {      // generated operand: VALU + ds_store
        int j  = kg / H;
        int kk = kg - j * H;
#pragma unroll
        for (int i = 0; i < 2; ++i) {
            int idx = tid + i * 256;
            int row = idx >> 2;
            int c8  = (idx & 3) * 8;
            size_t gr = (size_t)(tileM + row);
            v8bf cv = *reinterpret_cast<const v8bf*>(&Cbf[gr * H + kk + c8]);
            v8bf av;
#pragma unroll
            for (int e = 0; e < 8; ++e) {
                float c  = (float)cv[e];
                int   hc = kk + c8 + e;
                float val;
                if      (j == 0) val = c;
                else if (j == 1) val = c * Rb[hc];
                else if (j == 2) val = c * Pb[hc];
                else if (j == 3) val = fabsf(c - Rb[hc]);
                else             val = fabsf(c - Pb[hc]);
                av[e] = (bf16_t)val;
            }
            *reinterpret_cast<v8bf*>(&lsA[buf][row * LDS_STRIDE + c8]) = av;
        }
    };

    const int K = 5 * H;
    stageB(0, 0);
    stageA(0, 0);
    async_wait0();
    __syncthreads();

    int cur = 0;
    for (int kg = 0; kg < K; kg += TK) {
        if (kg + TK < K) stageB(kg + TK, cur ^ 1);     // async weight prefetch

        const int kb = (lane & 16) ? 8 : 0;
        const int rr = lane & 15;
        v16bf af[4], wf[4];
#pragma unroll
        for (int mi = 0; mi < 4; ++mi) af[mi] = frag_from_lds(lsA[cur], wm * 64 + mi * 16 + rr, kb);
#pragma unroll
        for (int ni = 0; ni < 4; ++ni) wf[ni] = frag_from_lds(lsB[cur], wn * 64 + ni * 16 + rr, kb);
#pragma unroll
        for (int mi = 0; mi < 4; ++mi)
#pragma unroll
            for (int ni = 0; ni < 4; ++ni)
                acc[mi][ni] = __builtin_amdgcn_wmma_f32_16x16x32_bf16(
                    false, af[mi], false, wf[ni], (short)0, acc[mi][ni], false, false);

        if (kg + TK < K) stageA(kg + TK, cur ^ 1);     // VALU co-executes with matrix pipe

        async_wait0();
        __syncthreads();
        cur ^= 1;
    }

    const int nlo = lane & 15;
    const int mhi = (lane >> 4) << 3;
#pragma unroll
    for (int mi = 0; mi < 4; ++mi) {
#pragma unroll
        for (int ni = 0; ni < 4; ++ni) {
            int   n    = tileN + wn * 64 + ni * 16 + nlo;
            float w8n  = w8[n];
            float w9n  = w9[n];
            float bsn  = basev[(size_t)b * H + n];
            long mbase = tileM + wm * 64 + mi * 16 + mhi;
#pragma unroll
            for (int r = 0; r < 8; ++r) {
                size_t gr = (size_t)(mbase + r);
                float  v  = acc[mi][ni][r] + bsn + sR[gr] * w8n + sP[gr] * w9n;
                Hid[gr * H + n] = (bf16_t)tanhf(v);
            }
        }
    }
}

// sR[row] = dot(C[row], aR[b]) ; sP[row] = dot(C[row], aP[b])    (one wave per row)
__global__ __launch_bounds__(256) void row_dot_kernel(
    float* __restrict__ sR, float* __restrict__ sP, const float* __restrict__ C,
    const float* __restrict__ aR, const float* __restrict__ aP, int S, int H)
{
    int row  = blockIdx.x * 8 + (threadIdx.x >> 5);
    int lane = threadIdx.x & 31;
    int b    = row / S;
    const float* cr = C  + (size_t)row * H;
    const float* ar = aR + (size_t)b * H;
    const float* ap = aP + (size_t)b * H;
    float s1 = 0.0f, s2 = 0.0f;
    for (int k = lane; k < H; k += 32) { float c = cr[k]; s1 = fmaf(c, ar[k], s1); s2 = fmaf(c, ap[k], s2); }
#pragma unroll
    for (int off = 16; off > 0; off >>= 1) { s1 += __shfl_xor(s1, off, 32); s2 += __shfl_xor(s2, off, 32); }
    if (lane == 0) { sR[row] = s1; sP[row] = s2; }
}

// g[row] = sigmoid(dot(Hid[row], gW2) + gb)
__global__ __launch_bounds__(256) void gate_reduce_kernel(
    float* __restrict__ g, const bf16_t* __restrict__ Hid,
    const float* __restrict__ gW2, const float* __restrict__ gW2b, int H)
{
    int row  = blockIdx.x * 8 + (threadIdx.x >> 5);
    int lane = threadIdx.x & 31;
    const bf16_t* hr = Hid + (size_t)row * H;
    float s = 0.0f;
    for (int k = lane; k < H; k += 32) s = fmaf((float)hr[k], gW2[k], s);
#pragma unroll
    for (int off = 16; off > 0; off >>= 1) s += __shfl_xor(s, off, 32);
    if (lane == 0) g[row] = sigmoidf_(s + gW2b[0]);
}

__global__ void f32_to_bf16_kernel(bf16_t* __restrict__ dst, const float* __restrict__ src, size_t n) {
    size_t i = (size_t)blockIdx.x * blockDim.x + threadIdx.x;
    if (i < n) dst[i] = (bf16_t)src[i];
}

// Repack gW1_W (H x Z, Z=7H+2) -> gW1p (H x 7H) bf16 + f32 columns w8, w9.
__global__ void repack_gw1_kernel(bf16_t* __restrict__ gW1p, float* __restrict__ w8,
                                  float* __restrict__ w9, const float* __restrict__ gW1,
                                  int H, int Z)
{
    int    H7 = 7 * H;
    size_t i  = (size_t)blockIdx.x * blockDim.x + threadIdx.x;
    size_t total = (size_t)H * (size_t)H7;
    if (i < total) {
        size_t nrow = i / H7;
        size_t c    = i - nrow * H7;
        gW1p[i] = (bf16_t)gW1[nrow * Z + c];
    }
    if (i < (size_t)H) {
        w8[i] = gW1[i * (size_t)Z + H7];
        w9[i] = gW1[i * (size_t)Z + H7 + 1];
    }
}

__global__ void zero_state_kernel(float* __restrict__ h, bf16_t* __restrict__ hb, size_t n) {
    size_t i = (size_t)blockIdx.x * blockDim.x + threadIdx.x;
    if (i < n) { h[i] = 0.0f; hb[i] = (bf16_t)0.0f; }
}

__global__ void init_state_kernel(float* __restrict__ M, bf16_t* __restrict__ Mb,
                                  const float* __restrict__ Q, size_t n) {
    size_t i = (size_t)blockIdx.x * blockDim.x + threadIdx.x;
    if (i < n) { float v = Q[i]; M[i] = v; Mb[i] = (bf16_t)v; }
}

// Episodic GRU step: h = g*GRU(c_t,h) + (1-g)*h  (gi precomputed incl. bias; gh incl. bias)
__global__ __launch_bounds__(256) void gru_step_kernel(
    float* __restrict__ h, bf16_t* __restrict__ hbf, const float* __restrict__ giAll,
    const float* __restrict__ gh, const float* __restrict__ gatt, int t, int S, int H)
{
    int i = blockIdx.x * 256 + threadIdx.x;   // 0 .. B*H-1
    int b = i / H;
    int j = i - b * H;
    const float* gi = giAll + ((size_t)b * S + t) * (size_t)(3 * H);
    const float* gb = gh + (size_t)b * 3 * H;
    float r  = sigmoidf_(gi[j] + gb[j]);
    float z  = sigmoidf_(gi[H + j] + gb[H + j]);
    float nn = tanhf(gi[2 * H + j] + r * gb[2 * H + j]);
    float hp = h[i];
    float hn = (1.0f - z) * nn + z * hp;
    float gt = gatt[(size_t)b * S + t];
    float o  = gt * hn + (1.0f - gt) * hp;
    h[i] = o;
    hbf[i] = (bf16_t)o;
}

// Memory GRU: M = GRU(h_T, M)
__global__ __launch_bounds__(256) void mem_gru_kernel(
    float* __restrict__ Mcur, bf16_t* __restrict__ Mbf, float* __restrict__ dout,
    const float* __restrict__ gi, const float* __restrict__ gh, int H, int writeOut)
{
    int i = blockIdx.x * 256 + threadIdx.x;
    int b = i / H;
    int j = i - b * H;
    const float* gib = gi + (size_t)b * 3 * H;
    const float* ghb = gh + (size_t)b * 3 * H;
    float r  = sigmoidf_(gib[j] + ghb[j]);
    float z  = sigmoidf_(gib[H + j] + ghb[H + j]);
    float nn = tanhf(gib[2 * H + j] + r * ghb[2 * H + j]);
    float hp = Mcur[i];
    float o  = (1.0f - z) * nn + z * hp;
    Mcur[i] = o;
    Mbf[i]  = (bf16_t)o;
    if (writeOut) dout[i] = o;
}

extern "C" void kernel_launch(void* const* d_in, const int* in_sizes, int n_in,
                              void* d_out, int out_size, void* d_ws, size_t ws_size,
                              hipStream_t stream)
{
    (void)in_sizes; (void)n_in; (void)out_size; (void)ws_size;
    const int Bb = 128, S = 256, H = 1024;
    const int BS = Bb * S, H3 = 3 * H, H7 = 7 * H, Z = 7 * H + 2;

    const float* C    = (const float*)d_in[0];
    const float* Q    = (const float*)d_in[1];
    const float* attW = (const float*)d_in[2];
    const float* gW1  = (const float*)d_in[3];
    const float* gW1b = (const float*)d_in[4];
    const float* gW2  = (const float*)d_in[5];
    const float* gW2b = (const float*)d_in[6];
    const float* WeI  = (const float*)d_in[7];
    const float* WeH  = (const float*)d_in[8];
    const float* beI  = (const float*)d_in[9];
    const float* beH  = (const float*)d_in[10];
    const float* WmI  = (const float*)d_in[11];
    const float* WmH  = (const float*)d_in[12];
    const float* bmI  = (const float*)d_in[13];
    const float* bmH  = (const float*)d_in[14];
    float* out = (float*)d_out;

    char*  w   = (char*)d_ws;
    size_t off = 0;
    auto alloc = [&](size_t bytes) -> char* {
        char* p = w + off;
        off += (bytes + 255) & ~(size_t)255;
        return p;
    };

    bf16_t* Cbf    = (bf16_t*)alloc((size_t)BS * H * 2);
    bf16_t* gW1p   = (bf16_t*)alloc((size_t)H * H7 * 2);
    float*  w8     = (float*) alloc((size_t)H * 4);
    float*  w9     = (float*) alloc((size_t)H * 4);
    bf16_t* attWbf = (bf16_t*)alloc((size_t)H * H * 2);
    bf16_t* WeIbf  = (bf16_t*)alloc((size_t)H3 * H * 2);
    bf16_t* WeHbf  = (bf16_t*)alloc((size_t)H3 * H * 2);
    bf16_t* WmIbf  = (bf16_t*)alloc((size_t)H3 * H * 2);
    bf16_t* WmHbf  = (bf16_t*)alloc((size_t)H3 * H * 2);
    bf16_t* Qbf    = (bf16_t*)alloc((size_t)Bb * H * 2);
    float*  giAll  = (float*) alloc((size_t)BS * H3 * 4);
    bf16_t* Hid    = (bf16_t*)alloc((size_t)BS * H * 2);
    float*  gatt   = (float*) alloc((size_t)BS * 4);
    float*  sR     = (float*) alloc((size_t)BS * 4);
    float*  sP     = (float*) alloc((size_t)BS * 4);
    float*  aR     = (float*) alloc((size_t)Bb * H * 4);
    float*  aP     = (float*) alloc((size_t)Bb * H * 4);
    float*  basev  = (float*) alloc((size_t)Bb * H * 4);
    float*  hst    = (float*) alloc((size_t)Bb * H * 4);
    bf16_t* hbf    = (bf16_t*)alloc((size_t)Bb * H * 2);
    float*  ghbuf  = (float*) alloc((size_t)Bb * H3 * 4);
    float*  gi2    = (float*) alloc((size_t)Bb * H3 * 4);
    float*  gh2    = (float*) alloc((size_t)Bb * H3 * 4);
    float*  Mcur   = (float*) alloc((size_t)Bb * H * 4);
    bf16_t* Mbf    = (bf16_t*)alloc((size_t)Bb * H * 2);

    auto cdiv = [](size_t a, size_t b) { return (unsigned)((a + b - 1) / b); };

    // ---- one-time conversions / precompute ----
    { size_t n = (size_t)BS * H; f32_to_bf16_kernel<<<cdiv(n,256),256,0,stream>>>(Cbf, C, n); }
    { size_t n = (size_t)H * H;  f32_to_bf16_kernel<<<cdiv(n,256),256,0,stream>>>(attWbf, attW, n); }
    { size_t n = (size_t)H3 * H;
      f32_to_bf16_kernel<<<cdiv(n,256),256,0,stream>>>(WeIbf, WeI, n);
      f32_to_bf16_kernel<<<cdiv(n,256),256,0,stream>>>(WeHbf, WeH, n);
      f32_to_bf16_kernel<<<cdiv(n,256),256,0,stream>>>(WmIbf, WmI, n);
      f32_to_bf16_kernel<<<cdiv(n,256),256,0,stream>>>(WmHbf, WmH, n); }
    { size_t n = (size_t)Bb * H; f32_to_bf16_kernel<<<cdiv(n,256),256,0,stream>>>(Qbf, Q, n); }
    { size_t n = (size_t)H * H7; repack_gw1_kernel<<<cdiv(n,256),256,0,stream>>>(gW1p, w8, w9, gW1, H, Z); }

    // giAll = C @ We_i.T + be_i   (big WMMA GEMM, reused by all 3 hops)
    gemm_bf16_kernel<<<dim3(H3/TN, BS/TM), 256, 0, stream>>>(giAll, Cbf, WeIbf, beI, H3, H, H, 0, 0);
    // aP = Q @ att_W.T (fixed across hops)
    gemm_bf16_kernel<<<dim3(H/TN, Bb/TM), 256, 0, stream>>>(aP, Qbf, attWbf, nullptr, H, H, H, 0, 0);
    // M <- Q
    { size_t n = (size_t)Bb * H; init_state_kernel<<<cdiv(n,256),256,0,stream>>>(Mcur, Mbf, Q, n); }

    for (int it = 0; it < 3; ++it) {
        // aR = M @ att_W.T
        gemm_bf16_kernel<<<dim3(H/TN, Bb/TM), 256, 0, stream>>>(aR, Mbf, attWbf, nullptr, H, H, H, 0, 0);
        // base = Q @ W1.T + M @ W2.T + gW1_b   (gW1 column blocks [H,2H) and [2H,3H))
        gemm_bf16_kernel<<<dim3(H/TN, Bb/TM), 256, 0, stream>>>(basev, Qbf, gW1p, gW1b, H, H, H7, H,     0);
        gemm_bf16_kernel<<<dim3(H/TN, Bb/TM), 256, 0, stream>>>(basev, Mbf, gW1p, nullptr, H, H, H7, 2*H, 1);
        // sR/sP row dots
        row_dot_kernel<<<BS/8, 256, 0, stream>>>(sR, sP, C, aR, aP, S, H);
        // fused K=5H gate GEMM -> Hid (tanh)
        fused_gate_gemm_kernel<<<dim3(H/TN, BS/TM), 256, 0, stream>>>(
            Hid, Cbf, Mcur, Q, gW1p, w8, w9, basev, sR, sP, S, H);
        // g = sigmoid(Hid . gW2 + b)
        gate_reduce_kernel<<<BS/8, 256, 0, stream>>>(gatt, Hid, gW2, gW2b, H);

        // episodic GRU scan over S
        { size_t n = (size_t)Bb * H; zero_state_kernel<<<cdiv(n,256),256,0,stream>>>(hst, hbf, n); }
        for (int t = 0; t < S; ++t) {
            gemm_bf16_kernel<<<dim3(H3/TN, Bb/TM), 256, 0, stream>>>(ghbuf, hbf, WeHbf, beH, H3, H, H, 0, 0);
            gru_step_kernel<<<(Bb*H)/256, 256, 0, stream>>>(hst, hbf, giAll, ghbuf, gatt, t, S, H);
        }

        // memory GRU: M = GRU(h_T, M)
        gemm_bf16_kernel<<<dim3(H3/TN, Bb/TM), 256, 0, stream>>>(gi2, hbf, WmIbf, bmI, H3, H, H, 0, 0);
        gemm_bf16_kernel<<<dim3(H3/TN, Bb/TM), 256, 0, stream>>>(gh2, Mbf, WmHbf, bmH, H3, H, H, 0, 0);
        mem_gru_kernel<<<(Bb*H)/256, 256, 0, stream>>>(Mcur, Mbf, out, gi2, gh2, H, (it == 2) ? 1 : 0);
    }
}
